// MatchLSTM_49658411877022
// MI455X (gfx1250) — compile-verified, matrix operand order checked by reading, not run on hardware
//
#include <hip/hip_runtime.h>

// MatchLSTM on gfx1250: all GEMMs via v_wmma_f32_16x16x32_bf16 (f32 accum),
// bf16-packed operands, double-buffered LDS 128x64 tiles (K staged 64 at a
// time, one workgroup barrier per K-iteration), softmax/LSTM cell in f32 VALU.

#define B_     256
#define TP_    512
#define TQ_    64
#define DP_    300
#define H_     256
#define KP_    320   // DP padded to multiple of 64
#define XK_    576   // xbuf row: ctx(320) | h(256)
#define KG_    896   // gates K: p(320) | ctx(320) | h(256)
#define FOURH_ 1024

typedef __attribute__((ext_vector_type(16))) __bf16 bf16x16;
typedef __attribute__((ext_vector_type(8)))  float  f32x8;

__device__ __forceinline__ unsigned short f2bf(float f) {
  unsigned int u = __float_as_uint(f);
  u += 0x7FFFu + ((u >> 16) & 1u);     // round-to-nearest-even
  return (unsigned short)(u >> 16);
}
__device__ __forceinline__ float sigm(float x) { return 1.0f / (1.0f + __expf(-x)); }

// ---------------------------------------------------------------------------
// Packing kernels (run once per launch)
// ---------------------------------------------------------------------------

// input_p (B,TP,DP) f32  ->  ppack [TP][B][KP_] bf16 (step-major, zero-padded)
__global__ __launch_bounds__(256) void pack_p(const float* __restrict__ input_p,
                                              unsigned short* __restrict__ ppack) {
  size_t gid = (size_t)blockIdx.x * blockDim.x + threadIdx.x;   // < TP*B*KP
  int k = (int)(gid % KP_);
  int b = (int)((gid / KP_) % B_);
  int t = (int)(gid / ((size_t)KP_ * B_));
  ppack[gid] = (k < DP_) ? f2bf(input_p[((size_t)b * TP_ + t) * DP_ + k])
                         : (unsigned short)0;
}

// input_q (B,TQ,DQ) f32 -> qpack [B*TQ][KP_] bf16
__global__ __launch_bounds__(256) void pack_q(const float* __restrict__ input_q,
                                              unsigned short* __restrict__ qpack) {
  size_t gid = (size_t)blockIdx.x * blockDim.x + threadIdx.x;   // < B*TQ*KP
  int k = (int)(gid % KP_);
  size_t row = gid / KP_;
  qpack[gid] = (k < DP_) ? f2bf(input_q[row * DP_ + k]) : (unsigned short)0;
}

// Build bf16 B^T matrices (row n holds the K-strip of output column n):
//   WqT    [H][KP_]      from Wq   (DP,H)
//   attBT  [H][XK_]      rows: Wp (k<300) | pad | Wr (320..575)
//   gatesBT[4H][KG_]     rows: W_ih[:, :300] | pad | W_ih[:,300:600] | pad | W_hh
__global__ __launch_bounds__(256) void pack_weights(
    const float* __restrict__ Wq, const float* __restrict__ Wp,
    const float* __restrict__ Wr, const float* __restrict__ W_ih,
    const float* __restrict__ W_hh,
    unsigned short* __restrict__ WqT, unsigned short* __restrict__ attBT,
    unsigned short* __restrict__ gatesBT) {
  int gid = blockIdx.x * blockDim.x + threadIdx.x;
  if (gid < H_ * KP_) {
    int n = gid / KP_, k = gid % KP_;
    WqT[gid] = (k < DP_) ? f2bf(Wq[k * H_ + n]) : (unsigned short)0;
  }
  if (gid < H_ * XK_) {
    int n = gid / XK_, k = gid % XK_;
    float v = 0.0f;
    if (k < DP_)            v = Wp[k * H_ + n];
    else if (k >= KP_)      v = Wr[(k - KP_) * H_ + n];
    attBT[gid] = f2bf(v);
  }
  if (gid < FOURH_ * KG_) {
    int n = gid / KG_, k = gid % KG_;
    float v = 0.0f;
    if (k < DP_)                       v = W_ih[n * 600 + k];             // p part
    else if (k >= KP_ && k < KP_+DP_)  v = W_ih[n * 600 + 300 + (k-KP_)]; // ctx part
    else if (k >= 2*KP_)               v = W_hh[n * H_ + (k - 2*KP_)];    // h part
    gatesBT[gid] = f2bf(v);
  }
}

// zero xbuf/h/c, precompute gate bias = b_ih + b_hh
__global__ __launch_bounds__(256) void init_state(
    unsigned short* __restrict__ xbuf, float* __restrict__ h, float* __restrict__ c,
    float* __restrict__ gbias, const float* __restrict__ b_ih,
    const float* __restrict__ b_hh) {
  int gid = blockIdx.x * blockDim.x + threadIdx.x;
  if (gid < B_ * XK_) xbuf[gid] = 0;
  if (gid < B_ * H_) { h[gid] = 0.0f; c[gid] = 0.0f; }
  if (gid < FOURH_)   gbias[gid] = b_ih[gid] + b_hh[gid];
}

// ---------------------------------------------------------------------------
// bf16 WMMA GEMM:  C[M,N] = concat(A0[:,0:K0], A1[:,0:Ktot-K0]) * B^T  (+bias)
// Block tile 128(M) x 64(N), 256 threads = 8 waves (4x2 of 32x32).
// K staged through double-buffered LDS in steps of 64 (two 32-wide WMMA
// subtiles per stage): ONE barrier per K-iteration, global loads of tile k+1
// overlap the 8 v_wmma of tile k.  K0 and Ktot must be multiples of 64.
// ---------------------------------------------------------------------------
#define TM 128
#define TN 64
#define TK 64
#define TKP 72   // padded LDS K-stride (bank stagger, 16B-aligned rows)

__global__ __launch_bounds__(256) void gemm_bf16_wmma(
    const unsigned short* __restrict__ A0, int lda0, int K0,
    const unsigned short* __restrict__ A1, int lda1,
    const unsigned short* __restrict__ BT, int ldb,
    const float* __restrict__ bias,
    float* __restrict__ C, int ldc, int Ktot) {
  __shared__ unsigned short As[2][TM * TKP];   // 2 x 18 KB
  __shared__ unsigned short Bs[2][TN * TKP];   // 2 x  9 KB

  const int tid  = threadIdx.x;
  const int lane = tid & 31;
  const int wave = tid >> 5;
  const int wm = (wave & 3) * 32;       // wave M offset inside tile
  const int wn = (wave >> 2) * 32;      // wave N offset inside tile
  const int m0 = blockIdx.x * TM;
  const int n0 = blockIdx.y * TN;

  const int lrow  = lane & 15;          // row (A) / col (B) within 16
  const int khalf = (lane >> 4) * 8;    // first K element of this lane's half

  uint4 ra[4], rb[2];                   // register staging for one K-tile

  // load one 128x64 A tile + 64x64 B tile into registers
  auto load_tile = [&](int kt) {
    const unsigned short* Asrc; int lda;
    if (kt < K0) { Asrc = A0 + kt;        lda = lda0; }
    else         { Asrc = A1 + (kt - K0); lda = lda1; }
#pragma unroll
    for (int i = 0; i < 4; ++i) {       // 1024 x 16B chunks, 4 per thread
      int chunk = tid + i * 256;
      int r  = chunk >> 3;
      int c8 = (chunk & 7) * 8;
      ra[i] = *reinterpret_cast<const uint4*>(&Asrc[(size_t)(m0 + r) * lda + c8]);
    }
#pragma unroll
    for (int i = 0; i < 2; ++i) {       // 512 x 16B chunks, 2 per thread
      int chunk = tid + i * 256;
      int r  = chunk >> 3;
      int c8 = (chunk & 7) * 8;
      rb[i] = *reinterpret_cast<const uint4*>(&BT[(size_t)(n0 + r) * ldb + kt + c8]);
    }
  };
  auto store_tile = [&](int buf) {
#pragma unroll
    for (int i = 0; i < 4; ++i) {
      int chunk = tid + i * 256;
      int r  = chunk >> 3;
      int c8 = (chunk & 7) * 8;
      *reinterpret_cast<uint4*>(&As[buf][r * TKP + c8]) = ra[i];
    }
#pragma unroll
    for (int i = 0; i < 2; ++i) {
      int chunk = tid + i * 256;
      int r  = chunk >> 3;
      int c8 = (chunk & 7) * 8;
      *reinterpret_cast<uint4*>(&Bs[buf][r * TKP + c8]) = rb[i];
    }
  };

  f32x8 acc[2][2] = {};

  load_tile(0);
  store_tile(0);
  __syncthreads();

  int cur = 0;
  for (int kt = 0; kt < Ktot; kt += TK) {
    const bool has_next = (kt + TK) < Ktot;
    if (has_next) {
      load_tile(kt + TK);               // global loads overlap WMMA below
      if (kt + 2 * TK < Ktot) {         // prefetch tile after next
        int nk = kt + 2 * TK;
        const unsigned short* nsrc = (nk < K0) ? (A0 + nk) : (A1 + (nk - K0));
        int nlda = (nk < K0) ? lda0 : lda1;
        __builtin_prefetch(&nsrc[(size_t)(m0 + (tid >> 3)) * nlda], 0, 0);
      }
    }

    const unsigned short* Asb = As[cur];
    const unsigned short* Bsb = Bs[cur];
#pragma unroll
    for (int ks = 0; ks < 2; ++ks) {    // two 32-wide K subtiles per stage
      union Frag { bf16x16 v; uint4 q[2]; };
      Frag a[2], b[2];
      const int ko = ks * 32 + khalf;
#pragma unroll
      for (int mi = 0; mi < 2; ++mi) {
        int row = wm + mi * 16 + lrow;
        a[mi].q[0] = *reinterpret_cast<const uint4*>(&Asb[row * TKP + ko]);
        a[mi].q[1] = *reinterpret_cast<const uint4*>(&Asb[row * TKP + ko + 16]);
      }
#pragma unroll
      for (int ni = 0; ni < 2; ++ni) {
        int col = wn + ni * 16 + lrow;
        b[ni].q[0] = *reinterpret_cast<const uint4*>(&Bsb[col * TKP + ko]);
        b[ni].q[1] = *reinterpret_cast<const uint4*>(&Bsb[col * TKP + ko + 16]);
      }
#pragma unroll
      for (int mi = 0; mi < 2; ++mi)
#pragma unroll
        for (int ni = 0; ni < 2; ++ni)
          acc[mi][ni] = __builtin_amdgcn_wmma_f32_16x16x32_bf16(
              false, a[mi].v, false, b[ni].v, (short)0, acc[mi][ni], false, false);
    }

    if (has_next) {
      store_tile(cur ^ 1);              // write OTHER buffer: no read/write clash
      __syncthreads();                  // single barrier per K-iteration
      cur ^= 1;
    }
  }

  // epilogue: VGPR r -> M = r + 8*(lane/16), N = lane%16
#pragma unroll
  for (int mi = 0; mi < 2; ++mi)
#pragma unroll
    for (int ni = 0; ni < 2; ++ni) {
      int n     = n0 + wn + ni * 16 + lrow;
      int mbase = m0 + wm + mi * 16 + ((lane >> 4) * 8);
      float bv = bias ? bias[n] : 0.0f;
#pragma unroll
      for (int r = 0; r < 8; ++r)
        C[(size_t)(mbase + r) * ldc + n] = acc[mi][ni][r] + bv;
    }
}

// ---------------------------------------------------------------------------
// Attention: g = tanh(q_proj + attv), scores = g . w_att, softmax over TQ,
// ctx = alpha @ input_q  ->  bf16 into xbuf[b][0:320)
// One block per batch element, 256 threads.
// ---------------------------------------------------------------------------
__global__ __launch_bounds__(256) void att_softmax_ctx(
    const float* __restrict__ qproj,    // [B*TQ][H]
    const float* __restrict__ attv,     // [B][H]  (includes b_att)
    const float* __restrict__ w_att,    // [H]
    const float* __restrict__ mask_q,   // [B][TQ]
    const float* __restrict__ input_q,  // [B*TQ][DP]
    unsigned short* __restrict__ xbuf)  // [B][XK_]
{
  __shared__ float sv[H_], sw[H_];
  __shared__ float spart[TQ_ * 4];
  __shared__ float ssc[TQ_];
  __shared__ float sinv;
  int b = blockIdx.x, tid = threadIdx.x;
  sv[tid] = attv[b * H_ + tid];
  sw[tid] = w_att[tid];
  __syncthreads();

  int tq = tid >> 2, part = tid & 3;
  const float* qp = qproj + ((size_t)b * TQ_ + tq) * H_ + part * 64;
  float s = 0.0f;
#pragma unroll 8
  for (int hh = 0; hh < 64; ++hh)
    s += tanhf(qp[hh] + sv[part * 64 + hh]) * sw[part * 64 + hh];
  spart[tq * 4 + part] = s;
  __syncthreads();

  if (tid < TQ_) {
    float sc = spart[tid*4] + spart[tid*4+1] + spart[tid*4+2] + spart[tid*4+3];
    if (!(mask_q[b * TQ_ + tid] > 0.0f)) sc = -1.0e9f;
    ssc[tid] = sc;
  }
  __syncthreads();
  if (tid == 0) {
    float mx = ssc[0];
    for (int i = 1; i < TQ_; ++i) mx = fmaxf(mx, ssc[i]);
    float sum = 0.0f;
    for (int i = 0; i < TQ_; ++i) { float e = __expf(ssc[i] - mx); ssc[i] = e; sum += e; }
    sinv = 1.0f / sum;
  }
  __syncthreads();
  float inv = sinv;

  for (int d = tid; d < KP_; d += 256) {
    float a = 0.0f;
    if (d < DP_) {
      const float* iq = input_q + (size_t)b * TQ_ * DP_ + d;
#pragma unroll 8
      for (int q = 0; q < TQ_; ++q) a += ssc[q] * iq[q * DP_];
      a *= inv;
    }
    xbuf[b * XK_ + d] = f2bf(a);
  }
}

// ---------------------------------------------------------------------------
// LSTM cell: gates -> (h,c) with mask blend; writes f32 h/c, bf16 h into xbuf,
// and the output slice out[b][t][:].
// ---------------------------------------------------------------------------
__global__ __launch_bounds__(256) void lstm_cell(
    const float* __restrict__ gates,   // [B][4H] (i|f|g|o)
    const float* __restrict__ mask_p,  // [B][TP]
    float* __restrict__ hbuf, float* __restrict__ cbuf,   // [B][H]
    unsigned short* __restrict__ xbuf,                    // h bf16 at [b][320+j]
    float* __restrict__ out, int t) {
  int b = blockIdx.x, j = threadIdx.x;
  const float* g = gates + (size_t)b * FOURH_;
  float gi = g[j], gf = g[H_ + j], gg = g[2 * H_ + j], go = g[3 * H_ + j];
  float c_old = cbuf[b * H_ + j], h_old = hbuf[b * H_ + j];
  float c_new = sigm(gf) * c_old + sigm(gi) * tanhf(gg);
  float h_new = sigm(go) * tanhf(c_new);
  float m = mask_p[(size_t)b * TP_ + t];
  h_new = h_new * m + h_old * (1.0f - m);
  c_new = c_new * m + c_old * (1.0f - m);
  cbuf[b * H_ + j] = c_new;
  hbuf[b * H_ + j] = h_new;
  xbuf[b * XK_ + KP_ + j] = f2bf(h_new);
  out[((size_t)b * TP_ + t) * H_ + j] = h_new;
}

// ---------------------------------------------------------------------------
extern "C" void kernel_launch(void* const* d_in, const int* in_sizes, int n_in,
                              void* d_out, int out_size, void* d_ws, size_t ws_size,
                              hipStream_t stream) {
  const float* input_p = (const float*)d_in[0];
  const float* mask_p  = (const float*)d_in[1];
  const float* input_q = (const float*)d_in[2];
  const float* mask_q  = (const float*)d_in[3];
  const float* Wq      = (const float*)d_in[4];
  const float* Wp      = (const float*)d_in[5];
  const float* Wr      = (const float*)d_in[6];
  const float* b_att   = (const float*)d_in[7];
  const float* w_att   = (const float*)d_in[8];
  const float* W_ih    = (const float*)d_in[9];
  const float* W_hh    = (const float*)d_in[10];
  const float* b_ih    = (const float*)d_in[11];
  const float* b_hh    = (const float*)d_in[12];
  float* out = (float*)d_out;

  char* ws = (char*)d_ws;
  size_t off = 0;
  auto take = [&](size_t bytes) -> char* {
    char* p = ws + off;
    off = (off + bytes + 255) & ~(size_t)255;
    return p;
  };
  unsigned short* ppack   = (unsigned short*)take((size_t)TP_ * B_ * KP_ * 2);
  unsigned short* qpack   = (unsigned short*)take((size_t)B_ * TQ_ * KP_ * 2);
  float*          qproj   = (float*)take((size_t)B_ * TQ_ * H_ * 4);
  unsigned short* WqT     = (unsigned short*)take((size_t)H_ * KP_ * 2);
  unsigned short* attBT   = (unsigned short*)take((size_t)H_ * XK_ * 2);
  unsigned short* gatesBT = (unsigned short*)take((size_t)FOURH_ * KG_ * 2);
  float*          attv    = (float*)take((size_t)B_ * H_ * 4);
  unsigned short* xbuf    = (unsigned short*)take((size_t)B_ * XK_ * 2);
  float*          gates   = (float*)take((size_t)B_ * FOURH_ * 4);
  float*          hbuf    = (float*)take((size_t)B_ * H_ * 4);
  float*          cbuf    = (float*)take((size_t)B_ * H_ * 4);
  float*          gbias   = (float*)take((size_t)FOURH_ * 4);

  // ---- one-time packing / precompute (parallel phase) ----
  pack_p<<<(TP_ * B_ * KP_) / 256, 256, 0, stream>>>(input_p, ppack);
  pack_q<<<(B_ * TQ_ * KP_) / 256, 256, 0, stream>>>(input_q, qpack);
  pack_weights<<<(FOURH_ * KG_) / 256, 256, 0, stream>>>(Wq, Wp, Wr, W_ih, W_hh,
                                                         WqT, attBT, gatesBT);
  init_state<<<(B_ * XK_ + 255) / 256, 256, 0, stream>>>(xbuf, hbuf, cbuf, gbias,
                                                         b_ih, b_hh);
  // q_proj = qpack @ Wq   (M=16384, N=256, K=320)
  gemm_bf16_wmma<<<dim3((B_ * TQ_) / TM, H_ / TN), 256, 0, stream>>>(
      qpack, KP_, KP_, nullptr, 0, WqT, KP_, nullptr, qproj, H_, KP_);

  // ---- recurrence (512 steps) ----
  for (int t = 0; t < TP_; ++t) {
    const unsigned short* pt = ppack + (size_t)t * B_ * KP_;
    // attv = [p_t | h] @ [Wp;Wr] + b_att   (M=256, N=256, K=576)
    gemm_bf16_wmma<<<dim3(B_ / TM, H_ / TN), 256, 0, stream>>>(
        pt, KP_, KP_, xbuf + KP_, XK_, attBT, XK_, b_att, attv, H_, XK_);
    // softmax + context -> xbuf[:, 0:320)
    att_softmax_ctx<<<B_, 256, 0, stream>>>(qproj, attv, w_att, mask_q, input_q, xbuf);
    // gates = [p_t | ctx | h] @ [W_ih;W_hh]^T + (b_ih+b_hh)  (M=256, N=1024, K=896)
    gemm_bf16_wmma<<<dim3(B_ / TM, FOURH_ / TN), 256, 0, stream>>>(
        pt, KP_, KP_, xbuf, XK_, gatesBT, KG_, gbias, gates, FOURH_, KG_);
    // LSTM pointwise; writes h (f32 + bf16 into xbuf) and output slice
    lstm_cell<<<B_, 256, 0, stream>>>(gates, mask_p, hbuf, cbuf, xbuf, out, t);
  }
}
